// localatt_62697932587137
// MI455X (gfx1250) — matrix operand order repaired and checked by reading, not compiled, not run on hardware
//
#include <hip/hip_runtime.h>
#include <hip/hip_bf16.h>
#include <cstdint>
#include <cstddef>

// Problem dims (fixed by the reference)
#define BB   32
#define TT   512
#define FDIM 256
#define NHID 1024
#define NCELL 512
#define H4   2048          // 4*NCELL
#define BT   (BB * TT)     // 16384
#define NOUT 8
#define HB_STRIDE 520      // 512 + 8 f16 pad to stagger LDS banks

#if __has_builtin(__builtin_amdgcn_global_load_async_to_lds_b128) && \
    __has_builtin(__builtin_amdgcn_s_wait_asynccnt)
#define HAVE_ASYNC_LDS 1
#else
#define HAVE_ASYNC_LDS 0
#endif

typedef __attribute__((ext_vector_type(16))) _Float16 v16h;
typedef __attribute__((ext_vector_type(8)))  _Float16 h8;
typedef __attribute__((ext_vector_type(8)))  float    v8f;
typedef int v4i __attribute__((vector_size(4 * sizeof(int))));
typedef __attribute__((address_space(3))) v4i lds_v4i;

static __device__ __forceinline__ v16h mk16(h8 lo, h8 hi) {
  v16h r;
#pragma unroll
  for (int i = 0; i < 8; ++i) { r[i] = lo[i]; r[i + 8] = hi[i]; }
  return r;
}

static __device__ __forceinline__ float wave_sum(float v) {
#pragma unroll
  for (int o = 16; o > 0; o >>= 1) v += __shfl_down(v, o, 32);
  return v;
}
static __device__ __forceinline__ float wave_max(float v) {
#pragma unroll
  for (int o = 16; o > 0; o >>= 1) v = fmaxf(v, __shfl_down(v, o, 32));
  return v;
}
static __device__ __forceinline__ float sigm(float x) {
  return 1.0f / (1.0f + __expf(-x));
}

// ---------------------------------------------------------------------------
// f32 -> f16 convert (grid-stride)
// ---------------------------------------------------------------------------
__global__ void cvt_f16_kernel(const float* __restrict__ s,
                               _Float16* __restrict__ d, size_t n) {
  size_t i = (size_t)blockIdx.x * blockDim.x + threadIdx.x;
  size_t stride = (size_t)gridDim.x * blockDim.x;
  for (; i < n; i += stride) d[i] = (_Float16)s[i];
}

__global__ void bias_sum_kernel(const float* __restrict__ a,
                                const float* __restrict__ b,
                                float* __restrict__ o, int n) {
  int i = blockIdx.x * blockDim.x + threadIdx.x;
  if (i < n) o[i] = a[i] + b[i];
}

// ---------------------------------------------------------------------------
// C[M,N] = A16[M,K] @ B16[N,K]^T + bias[N]   (WMMA f16 -> f32)
// Block: 256 thr = 8 waves arranged 2(M) x 4(N-groups).
// Wave tile: 16(M) x 128(N) -> 8 f32 accumulators, A reused 8x per k-step.
// Ping-pong double buffering (k unrolled by 2): loads land directly in the
// fragment registers consumed by the WMMAs -> no cross-iteration reg copies.
// Bias slice for this block staged in LDS via async global->LDS (ASYNCcnt).
// Grid: (N/512, M/32).  K must be a multiple of 64 (256/1024 here).
// ---------------------------------------------------------------------------
__global__ __launch_bounds__(256) void gemm_f16(
    const _Float16* __restrict__ A, const _Float16* __restrict__ Bm,
    const float* __restrict__ bias, float* __restrict__ C,
    int M, int N, int K) {
  const int wave = threadIdx.x >> 5, lane = threadIdx.x & 31;
  const int wm = wave & 1, wn = wave >> 1;
  const int row0 = blockIdx.y * 32 + wm * 16;
  const int col0 = blockIdx.x * 512 + wn * 128;
  const int l   = lane & 15;
  const int kbA = (lane >> 4) << 3;   // A: half-wave K-chunks at +0/+8, +16/+24
  const int kbB = (lane >> 4) << 3;   // B: half-wave K-chunks at +0/+8, +16/+24
  const int mref = (lane >> 4) << 3;  // C/D: M offset 0 or 8
  const _Float16* ap = A  + (size_t)(row0 + l) * K + kbA;
  const _Float16* bp = Bm + (size_t)(col0 + l) * K + kbB;

  __shared__ float biasLds[512];
#if HAVE_ASYNC_LDS
  if (bias != nullptr && threadIdx.x < 128) {
    // 128 lanes x 16B = 2 KB = the 512-float bias slice for this block
    const float* gsrc = bias + (size_t)blockIdx.x * 512 + threadIdx.x * 4;
    float* ldst = biasLds + threadIdx.x * 4;
    __builtin_amdgcn_global_load_async_to_lds_b128(
        (v4i*)gsrc, (lds_v4i*)ldst, 0, 0);
  }
#else
  if (bias != nullptr) {
    for (int i = threadIdx.x; i < 512; i += 256)
      biasLds[i] = bias[(size_t)blockIdx.x * 512 + i];
  }
#endif

  auto loadA = [&](int k) -> v16h {
    return mk16(*(const h8*)(ap + k), *(const h8*)(ap + k + 16));
  };
  auto loadB = [&](int j, int k) -> v16h {
    const _Float16* bj = bp + (size_t)j * 16 * K + k;
    return mk16(*(const h8*)bj, *(const h8*)(bj + 16));
  };

  v8f acc[8] = {};
  v16h a0, a1, b0[8], b1[8];
  // prologue: fragment set 0 for k = 0
  a0 = loadA(0);
#pragma unroll
  for (int j = 0; j < 8; ++j) b0[j] = loadB(j, 0);

  for (int k = 0; k < K; k += 64) {
    // issue loads for k+32 into set 1
    __builtin_prefetch(ap + k + 160, 0, 3);   // global_prefetch_b8
    a1 = loadA(k + 32);
#pragma unroll
    for (int j = 0; j < 8; ++j) b1[j] = loadB(j, k + 32);
    // consume set 0 (k)
#pragma unroll
    for (int j = 0; j < 8; ++j)
      acc[j] = __builtin_amdgcn_wmma_f32_16x16x32_f16(
          false, a0, false, b0[j], (short)0, acc[j], false, false);
    // issue loads for k+64 back into set 0
    if (k + 64 < K) {
      a0 = loadA(k + 64);
#pragma unroll
      for (int j = 0; j < 8; ++j) b0[j] = loadB(j, k + 64);
    }
    // consume set 1 (k+32)
#pragma unroll
    for (int j = 0; j < 8; ++j)
      acc[j] = __builtin_amdgcn_wmma_f32_16x16x32_f16(
          false, a1, false, b1[j], (short)0, acc[j], false, false);
  }

#if HAVE_ASYNC_LDS
  if (bias != nullptr && threadIdx.x < 128)
    __builtin_amdgcn_s_wait_asynccnt(0);
#endif
  __syncthreads();   // publish biasLds to all waves

#pragma unroll
  for (int j = 0; j < 8; ++j) {
    const int cloc = wn * 128 + j * 16 + l;          // column within block
    const float bv = (bias != nullptr) ? biasLds[cloc] : 0.0f;
    float* cp = C + (size_t)(row0 + mref) * N + blockIdx.x * 512 + cloc;
#pragma unroll
    for (int r = 0; r < 8; ++r) cp[(size_t)r * N] = acc[j][r] + bv;
  }
}

// ---------------------------------------------------------------------------
// LayerNorm over NHID + LeakyReLU(0.01), f32 in -> f16 out; optional pad mask
// One 256-thread block per row (BT rows).
// ---------------------------------------------------------------------------
__global__ __launch_bounds__(256) void ln_act_kernel(
    const float* __restrict__ X, const float* __restrict__ g,
    const float* __restrict__ be, _Float16* __restrict__ O,
    const int* __restrict__ lens) {
  const int row = blockIdx.x;
  const float* x = X + (size_t)row * NHID;
  __shared__ float red[8];
  const int tid = threadIdx.x, wave = tid >> 5, lane = tid & 31;

  float s = 0.0f;
  for (int i = tid; i < NHID; i += 256) s += x[i];
  s = wave_sum(s);
  if (lane == 0) red[wave] = s;
  __syncthreads();
  if (wave == 0) { float v = (lane < 8) ? red[lane] : 0.0f; v = wave_sum(v);
                   if (lane == 0) red[0] = v; }
  __syncthreads();
  const float mean = red[0] * (1.0f / NHID);
  __syncthreads();

  float var = 0.0f;
  for (int i = tid; i < NHID; i += 256) { float d = x[i] - mean; var += d * d; }
  var = wave_sum(var);
  if (lane == 0) red[wave] = var;
  __syncthreads();
  if (wave == 0) { float v = (lane < 8) ? red[lane] : 0.0f; v = wave_sum(v);
                   if (lane == 0) red[0] = v; }
  __syncthreads();
  const float rstd = rsqrtf(red[0] * (1.0f / NHID) + 1e-5f);

  bool pad = false;
  if (lens) { const int b = row / TT, t = row % TT; pad = (t >= lens[b]); }
  for (int i = tid; i < NHID; i += 256) {
    float z = (x[i] - mean) * rstd * g[i] + be[i];
    z = (z >= 0.0f) ? z : 0.01f * z;
    O[(size_t)row * NHID + i] = pad ? (_Float16)0.0f : (_Float16)z;
  }
}

// ---------------------------------------------------------------------------
// One bidirectional LSTM layer recurrence. grid = 2 blocks (dir), 1024 thr.
// Dynamic LDS: h16 state [32][HB_STRIDE] f16 (33 KB) + gates [32][2048] f32
// (256 KB) -> 289 KB, legal on CDNA5's 320 KB/WGP.
// Per step: gates = h16 @ Whh^T via 32 waves x 8 WMMA tiles, barrier, fused
// LSTM cell update (c state in registers), h written to hcat + LDS, barrier.
// ---------------------------------------------------------------------------
__global__ __launch_bounds__(1024) void lstm_layer_kernel(
    const float* __restrict__ xpF, const float* __restrict__ xpB,
    const _Float16* __restrict__ whh,   // layer base: [2][H4][NCELL] f16
    float* __restrict__ hcat) {         // [BT][2*NCELL] f32
  const int dir = blockIdx.x;
  const float* xp = dir ? xpB : xpF;
  const _Float16* Wh = whh + (size_t)dir * H4 * NCELL;
  extern __shared__ char smem[];
  _Float16* hbuf = (_Float16*)smem;                                  // 32 x 520
  float* gl = (float*)(smem + (size_t)BB * HB_STRIDE * sizeof(_Float16)); // 32 x 2048

  const int tid = threadIdx.x;
  const int wave = tid >> 5, lane = tid & 31;
  const int l   = lane & 15;
  const int kbA = (lane >> 4) << 3;
  const int kbB = (lane >> 4) << 3;
  const int mref = (lane >> 4) << 3;
  const int wm = wave & 1, wn = wave >> 1;     // 2(M) x 16(N-groups)
  const int row0 = wm * 16, col0 = wn * 128;

  for (int i = tid; i < BB * HB_STRIDE; i += 1024) hbuf[i] = (_Float16)0.0f;
  float c[16];
#pragma unroll
  for (int j = 0; j < 16; ++j) c[j] = 0.0f;
  __syncthreads();

  const _Float16* hp  = hbuf + (row0 + l) * HB_STRIDE + kbA;
  const _Float16* bp0 = Wh + (size_t)(col0 + l) * NCELL + kbB;

  for (int t = 0; t < TT; ++t) {
    const int ta = dir ? (TT - 1 - t) : t;
    // ---- recurrent GEMM: gates = h @ Whh^T ----
    v8f acc[8] = {};
    for (int k = 0; k < NCELL; k += 32) {
      v16h a = mk16(*(const h8*)(hp + k), *(const h8*)(hp + k + 16));
      v16h bts[8];
#pragma unroll
      for (int j = 0; j < 8; ++j) {     // issue all 8 tile loads first
        const _Float16* bj = bp0 + (size_t)j * 16 * NCELL + k;
        bts[j] = mk16(*(const h8*)bj, *(const h8*)(bj + 16));
      }
#pragma unroll
      for (int j = 0; j < 8; ++j)       // then 8 back-to-back WMMAs
        acc[j] = __builtin_amdgcn_wmma_f32_16x16x32_f16(
            false, a, false, bts[j], (short)0, acc[j], false, false);
    }
#pragma unroll
    for (int j = 0; j < 8; ++j) {
      float* gp = gl + (size_t)(row0 + mref) * H4 + col0 + j * 16 + l;
#pragma unroll
      for (int r = 0; r < 8; ++r) gp[(size_t)r * H4] = acc[j][r];
    }
    __syncthreads();
    // ---- fused cell update: each thread owns 16 (b,h) cells, c in regs ----
#pragma unroll
    for (int j = 0; j < 16; ++j) {
      const int e = j * 1024 + tid;
      const int bb = e >> 9, h = e & (NCELL - 1);
      const size_t xr = ((size_t)bb * TT + ta) * H4;
      const float gi = gl[bb * H4 + h]             + xp[xr + h];
      const float gf = gl[bb * H4 + NCELL + h]     + xp[xr + NCELL + h];
      const float gg = gl[bb * H4 + 2 * NCELL + h] + xp[xr + 2 * NCELL + h];
      const float go = gl[bb * H4 + 3 * NCELL + h] + xp[xr + 3 * NCELL + h];
      c[j] = sigm(gf) * c[j] + sigm(gi) * tanhf(gg);
      const float hv = sigm(go) * tanhf(c[j]);
      hcat[((size_t)bb * TT + ta) * (2 * NCELL) + dir * NCELL + h] = hv;
      hbuf[bb * HB_STRIDE + h] = (_Float16)hv;
    }
    __syncthreads();
  }
}

// ---------------------------------------------------------------------------
// Masked softmax attention pooling + classifier. One block per batch row.
// ---------------------------------------------------------------------------
__global__ __launch_bounds__(256) void attn_kernel(
    const float* __restrict__ H, const int* __restrict__ lens,
    const float* __restrict__ wu, const float* __restrict__ bu,
    const float* __restrict__ Wc, const float* __restrict__ bc,
    float* __restrict__ out) {
  const int b = blockIdx.x;
  __shared__ float sc[TT];
  __shared__ float pooled[2 * NCELL];
  __shared__ float red[8];
  const int tid = threadIdx.x, wave = tid >> 5, lane = tid & 31;
  const int len = lens[b];
  const float* Hb = H + (size_t)b * TT * (2 * NCELL);

  // scores
  for (int t = wave; t < TT; t += 8) {
    const float* h = Hb + (size_t)t * (2 * NCELL);
    float s = 0.0f;
    for (int i = lane; i < 2 * NCELL; i += 32) s += h[i] * wu[i];
    s = wave_sum(s);
    if (lane == 0) sc[t] = s + bu[0];
  }
  __syncthreads();
  // max over valid region
  float m = -3.0e38f;
  for (int t = tid; t < len; t += 256) m = fmaxf(m, sc[t]);
  m = wave_max(m);
  if (lane == 0) red[wave] = m;
  __syncthreads();
  if (wave == 0) { float v = (lane < 8) ? red[lane] : -3.0e38f; v = wave_max(v);
                   if (lane == 0) red[0] = v; }
  __syncthreads();
  m = red[0];
  __syncthreads();
  // exp + sum
  float se = 0.0f;
  for (int t = tid; t < len; t += 256) { float e = __expf(sc[t] - m); sc[t] = e; se += e; }
  se = wave_sum(se);
  if (lane == 0) red[wave] = se;
  __syncthreads();
  if (wave == 0) { float v = (lane < 8) ? red[lane] : 0.0f; v = wave_sum(v);
                   if (lane == 0) red[0] = v; }
  __syncthreads();
  const float inv = 1.0f / red[0];
  // weighted sum
  for (int d = tid; d < 2 * NCELL; d += 256) {
    float p = 0.0f;
    for (int t = 0; t < len; ++t) p += sc[t] * Hb[(size_t)t * (2 * NCELL) + d];
    pooled[d] = p * inv;
  }
  __syncthreads();
  if (tid < NOUT) {
    float s = bc[tid];
    const float* w = Wc + tid * (2 * NCELL);
    for (int i = 0; i < 2 * NCELL; ++i) s += pooled[i] * w[i];
    out[b * NOUT + tid] = s;
  }
}

// ---------------------------------------------------------------------------
extern "C" void kernel_launch(void* const* d_in, const int* in_sizes, int n_in,
                              void* d_out, int out_size, void* d_ws, size_t ws_size,
                              hipStream_t stream) {
  (void)in_sizes; (void)n_in; (void)out_size; (void)ws_size;
  const float* x   = (const float*)d_in[0];
  const int*   lens= (const int*)d_in[1];
  const float* W1  = (const float*)d_in[2];
  const float* b1  = (const float*)d_in[3];
  const float* g1  = (const float*)d_in[4];
  const float* be1 = (const float*)d_in[5];
  const float* W2  = (const float*)d_in[6];
  const float* b2  = (const float*)d_in[7];
  const float* g2  = (const float*)d_in[8];
  const float* be2 = (const float*)d_in[9];
  const float* wih = (const float*)d_in[10];  // [2,2,2048,1024]
  const float* whh = (const float*)d_in[11];  // [2,2,2048,512]
  const float* bih = (const float*)d_in[12];  // [2,2,2048]
  const float* bhh = (const float*)d_in[13];
  const float* wu  = (const float*)d_in[14];
  const float* bu  = (const float*)d_in[15];
  const float* Wc  = (const float*)d_in[16];
  const float* bc  = (const float*)d_in[17];
  float* out = (float*)d_out;

  char* ws = (char*)d_ws;
  size_t off = 0;
  auto take = [&](size_t bytes) -> char* {
    char* p = ws + off; off = (off + bytes + 255) & ~(size_t)255; return p;
  };
  _Float16* x16   = (_Float16*)take((size_t)BT * FDIM * 2);     // 8 MB
  _Float16* w1h   = (_Float16*)take((size_t)NHID * FDIM * 2);   // 0.5 MB
  _Float16* w2h   = (_Float16*)take((size_t)NHID * NHID * 2);   // 2 MB
  _Float16* wihh  = (_Float16*)take((size_t)4 * H4 * NHID * 2); // 16 MB
  _Float16* whhh  = (_Float16*)take((size_t)4 * H4 * NCELL * 2);// 8 MB
  float*    bsum  = (float*)take((size_t)4 * H4 * 4);           // 32 KB
  float*    Cbuf  = (float*)take((size_t)BT * NHID * 4);        // 64 MB (reused)
  _Float16* act16 = (_Float16*)take((size_t)BT * NHID * 2);     // 32 MB (reused)
  float*    xpF   = (float*)take((size_t)BT * H4 * 4);          // 128 MB
  float*    xpB   = (float*)take((size_t)BT * H4 * 4);          // 128 MB

  auto cvt = [&](const float* s, _Float16* d, size_t n) {
    int blocks = (int)((n + 255) / 256); if (blocks > 4096) blocks = 4096;
    cvt_f16_kernel<<<blocks, 256, 0, stream>>>(s, d, n);
  };
  cvt(x,   x16,  (size_t)BT * FDIM);
  cvt(W1,  w1h,  (size_t)NHID * FDIM);
  cvt(W2,  w2h,  (size_t)NHID * NHID);
  cvt(wih, wihh, (size_t)4 * H4 * NHID);
  cvt(whh, whhh, (size_t)4 * H4 * NCELL);
  bias_sum_kernel<<<(4 * H4 + 255) / 256, 256, 0, stream>>>(bih, bhh, bsum, 4 * H4);

  // MLP1: h1 = lrelu(LN(x @ W1^T + b1))
  gemm_f16<<<dim3(NHID / 512, BT / 32), 256, 0, stream>>>(x16, w1h, b1, Cbuf, BT, NHID, FDIM);
  ln_act_kernel<<<BT, 256, 0, stream>>>(Cbuf, g1, be1, act16, nullptr);
  // MLP2: h2 = mask * lrelu(LN(h1 @ W2^T + b2))
  gemm_f16<<<dim3(NHID / 512, BT / 32), 256, 0, stream>>>(act16, w2h, b2, Cbuf, BT, NHID, NHID);
  ln_act_kernel<<<BT, 256, 0, stream>>>(Cbuf, g2, be2, act16, lens);

  const size_t SMEM = (size_t)BB * HB_STRIDE * 2 + (size_t)BB * H4 * 4; // ~289 KB
  (void)hipFuncSetAttribute((const void*)lstm_layer_kernel,
                            hipFuncAttributeMaxDynamicSharedMemorySize, (int)SMEM);

  // LSTM layer 1 (input = h2 f16): xp = h2 @ wih^T + (bih+bhh), per dir
  gemm_f16<<<dim3(H4 / 512, BT / 32), 256, 0, stream>>>(act16, wihh,                      bsum,          xpF, BT, H4, NHID);
  gemm_f16<<<dim3(H4 / 512, BT / 32), 256, 0, stream>>>(act16, wihh + (size_t)H4 * NHID,  bsum + H4,     xpB, BT, H4, NHID);
  lstm_layer_kernel<<<2, 1024, SMEM, stream>>>(xpF, xpB, whhh, Cbuf);
  cvt(Cbuf, act16, (size_t)BT * NHID);   // hcat1 -> f16 for layer-2 projections

  // LSTM layer 2
  gemm_f16<<<dim3(H4 / 512, BT / 32), 256, 0, stream>>>(act16, wihh + (size_t)2 * H4 * NHID, bsum + 2 * H4, xpF, BT, H4, NHID);
  gemm_f16<<<dim3(H4 / 512, BT / 32), 256, 0, stream>>>(act16, wihh + (size_t)3 * H4 * NHID, bsum + 3 * H4, xpB, BT, H4, NHID);
  lstm_layer_kernel<<<2, 1024, SMEM, stream>>>(xpF, xpB, whhh + (size_t)2 * H4 * NCELL, Cbuf);

  // attention pooling + classifier
  attn_kernel<<<BB, 256, 0, stream>>>(Cbuf, lens, wu, bu, Wc, bc, out);
}
